// BiLSTM_CRF_25709674233911
// MI455X (gfx1250) — matrix operand order repaired and checked by reading, not compiled, not run on hardware
//
#include <hip/hip_runtime.h>
#include <hip/hip_bf16.h>

// ---------------------------------------------------------------------------
// BiLSTM-CRF on gfx1250 (MI455X). bf16 WMMA for the LSTM matmuls, f32 state.
// ---------------------------------------------------------------------------

typedef __bf16 bf16_t;
typedef __attribute__((ext_vector_type(16))) __bf16 v16bf;
typedef __attribute__((ext_vector_type(8)))  __bf16 v8bf;
typedef __attribute__((ext_vector_type(8)))  float  v8f;

#define SEQ    4096
#define EMBED  300
#define EPAD   320      // EMBED padded to multiple of 32 (WMMA K)
#define HID    512
#define G4H    2048     // 4*HID
#define TAGS   12
#define START_TAG 10
#define STOP_TAG  11
#define NEGINF (-10000.0f)

__device__ __forceinline__ bf16_t f32_to_bf16(float f) {
    union { float f; unsigned u; } v; v.f = f;
    unsigned r = v.u + 0x7FFFu + ((v.u >> 16) & 1u);   // round-to-nearest-even
    union { unsigned short s; bf16_t b; } o;
    o.s = (unsigned short)(r >> 16);
    return o.b;
}

__device__ __forceinline__ float sigmoidf_fast(float x) {
    return 1.0f / (1.0f + __expf(-x));
}

__device__ __forceinline__ v16bf load_a_frag(const bf16_t* row, int k0, int kg) {
    v8bf lo = *(const v8bf*)(row + k0 + 2 * kg);
    v8bf hv = *(const v8bf*)(row + k0 + 2 * kg + 8);
    v16bf a;
    #pragma unroll
    for (int e = 0; e < 8; ++e) { a[e] = lo[e]; a[8 + e] = hv[e]; }
    return a;
}

// ---------------------------------------------------------------------------
// 1) Embedding gather + f32->bf16 + K-pad to EPAD
// ---------------------------------------------------------------------------
__global__ __launch_bounds__(EPAD)
void k_embed(const int* __restrict__ sent, const float* __restrict__ table,
             bf16_t* __restrict__ emb) {
    int t = blockIdx.x;           // 0..SEQ-1
    int k = threadIdx.x;          // 0..EPAD-1
    int id = sent[t];
    float v = (k < EMBED) ? table[(long)id * EMBED + k] : 0.0f;
    emb[(long)t * EPAD + k] = f32_to_bf16(v);
}

// ---------------------------------------------------------------------------
// 2) Generic f32 -> bf16 convert with column pad (keeps [rows, cout] layout)
// ---------------------------------------------------------------------------
__global__ void k_cvt(const float* __restrict__ src, bf16_t* __restrict__ dst,
                      int rows, int cin, int cout) {
    long i = (long)blockIdx.x * blockDim.x + threadIdx.x;
    long total = (long)rows * cout;
    if (i >= total) return;
    int r = (int)(i / cout), c = (int)(i % cout);
    float v = (c < cin) ? src[(long)r * cin + c] : 0.0f;
    dst[i] = f32_to_bf16(v);
}

// ---------------------------------------------------------------------------
// 3) bias = bih + bhh
// ---------------------------------------------------------------------------
__global__ void k_bias(const float* __restrict__ a, const float* __restrict__ b,
                       float* __restrict__ o, int n) {
    int i = blockIdx.x * blockDim.x + threadIdx.x;
    if (i < n) o[i] = a[i] + b[i];
}

// ---------------------------------------------------------------------------
// 4) Input projection GEMM: xg[T,4H] = emb[T,EPAD] @ W[4H,EPAD]^T + bias
//    Each wave owns a 32(M) x 64(N) macro-tile = 2x4 WMMA tiles:
//    8 independent accumulation chains, A reused 4x, B reused 2x.
//    block = 128 (4 waves, each a different 64-wide N strip)
//    grid  = (G4H/256, SEQ/32, 2 dirs)
// ---------------------------------------------------------------------------
__global__ __launch_bounds__(128)
void k_gemm_xg(const bf16_t* __restrict__ emb,
               const bf16_t* __restrict__ wf, const bf16_t* __restrict__ wb,
               const float* __restrict__ biasf, const float* __restrict__ biasb,
               float* __restrict__ xgf, float* __restrict__ xgb) {
    const int lane = threadIdx.x & 31;
    const int wave = threadIdx.x >> 5;
    const int col0 = blockIdx.x * 256 + wave * 64;   // N strip (4 tiles)
    const int row0 = blockIdx.y * 32;                // M strip (2 tiles)
    const bf16_t* W    = blockIdx.z ? wb    : wf;
    const float*  bias = blockIdx.z ? biasb : biasf;
    float*        xg   = blockIdx.z ? xgb   : xgf;

    const int n  = lane & 15;
    const int hi = lane >> 4;          // half-wave select
    const int kg = hi * 8;             // A fragment K-group offset
    const int kh = hi * 16;            // B fragment K-half offset

    // C init = bias broadcast (bias depends only on N)
    v8f acc[2][4];
    #pragma unroll
    for (int nt = 0; nt < 4; ++nt) {
        float bn = bias[col0 + nt * 16 + n];
        #pragma unroll
        for (int r = 0; r < 8; ++r) { acc[0][nt][r] = bn; acc[1][nt][r] = bn; }
    }

    const bf16_t* pa0 = emb + (long)(row0 +  0 + n) * EPAD;   // A rows, M-tile 0
    const bf16_t* pa1 = emb + (long)(row0 + 16 + n) * EPAD;   // A rows, M-tile 1
    const bf16_t* pb0 = W + (long)(col0 +  0 + n) * EPAD + kh;
    const bf16_t* pb1 = W + (long)(col0 + 16 + n) * EPAD + kh;
    const bf16_t* pb2 = W + (long)(col0 + 32 + n) * EPAD + kh;
    const bf16_t* pb3 = W + (long)(col0 + 48 + n) * EPAD + kh;

    #pragma unroll
    for (int k0 = 0; k0 < EPAD; k0 += 32) {
        v16bf a0 = load_a_frag(pa0, k0, kg);
        v16bf a1 = load_a_frag(pa1, k0, kg);
        v16bf b0 = *(const v16bf*)(pb0 + k0);
        v16bf b1 = *(const v16bf*)(pb1 + k0);
        v16bf b2 = *(const v16bf*)(pb2 + k0);
        v16bf b3 = *(const v16bf*)(pb3 + k0);
        acc[0][0] = __builtin_amdgcn_wmma_f32_16x16x32_bf16(false, a0, false, b0, (short)0, acc[0][0], false, false);
        acc[0][1] = __builtin_amdgcn_wmma_f32_16x16x32_bf16(false, a0, false, b1, (short)0, acc[0][1], false, false);
        acc[0][2] = __builtin_amdgcn_wmma_f32_16x16x32_bf16(false, a0, false, b2, (short)0, acc[0][2], false, false);
        acc[0][3] = __builtin_amdgcn_wmma_f32_16x16x32_bf16(false, a0, false, b3, (short)0, acc[0][3], false, false);
        acc[1][0] = __builtin_amdgcn_wmma_f32_16x16x32_bf16(false, a1, false, b0, (short)0, acc[1][0], false, false);
        acc[1][1] = __builtin_amdgcn_wmma_f32_16x16x32_bf16(false, a1, false, b1, (short)0, acc[1][1], false, false);
        acc[1][2] = __builtin_amdgcn_wmma_f32_16x16x32_bf16(false, a1, false, b2, (short)0, acc[1][2], false, false);
        acc[1][3] = __builtin_amdgcn_wmma_f32_16x16x32_bf16(false, a1, false, b3, (short)0, acc[1][3], false, false);
    }

    #pragma unroll
    for (int mt = 0; mt < 2; ++mt) {
        #pragma unroll
        for (int r = 0; r < 8; ++r) {
            int m = row0 + mt * 16 + r + 8 * hi;
            float* dst = xg + (long)m * G4H + col0 + n;
            #pragma unroll
            for (int nt = 0; nt < 4; ++nt) dst[nt * 16] = acc[mt][nt][r];
        }
    }
}

// ---------------------------------------------------------------------------
// 5) Recurrent LSTM, one direction per block (grid=2), 1024 threads = 32 waves.
//    Per step: h(bf16, LDS) broadcast into A; each wave owns 4 output tiles
//    of 16 gates; B streamed from L2-resident Whh[4H,512] bf16.
//    Next step's xg row is prefetched under the WMMA matvec.
// ---------------------------------------------------------------------------
__global__ __launch_bounds__(1024)
void k_lstm(const float* __restrict__ xgf, const float* __restrict__ xgb,
            const bf16_t* __restrict__ whhf, const bf16_t* __restrict__ whhb,
            float* __restrict__ hcat) {
    __shared__ float  hf[HID];
    __shared__ bf16_t hb[HID];
    __shared__ float  g[G4H];

    const int tid  = threadIdx.x;
    const int lane = tid & 31;
    const int wave = tid >> 5;
    const int dir  = blockIdx.x;

    const float*  xg  = dir ? xgb  : xgf;
    const bf16_t* whh = dir ? whhb : whhf;

    const int n  = lane & 15;
    const int hi = lane >> 4;
    const int kg = hi * 8;
    const int kh = hi * 16;
    const int colbase = wave * 64;   // 4 tiles x 16 gates per wave

    const bf16_t* pb0 = whh + (long)(colbase +  0 + n) * HID + kh;
    const bf16_t* pb1 = whh + (long)(colbase + 16 + n) * HID + kh;
    const bf16_t* pb2 = whh + (long)(colbase + 32 + n) * HID + kh;
    const bf16_t* pb3 = whh + (long)(colbase + 48 + n) * HID + kh;

    float c = 0.0f;
    if (tid < HID) hf[tid] = 0.0f;
    __syncthreads();

    for (int s = 0; s < SEQ; ++s) {
        const int t = dir ? (SEQ - 1 - s) : s;

        // --- convert current h to bf16 for the matrix pipe ---
        if (tid < HID) hb[tid] = f32_to_bf16(hf[tid]);

        // --- prefetch next timestep's xg row (8KB = 64 x 128B lines) ---
        {
            int tn = dir ? (t - 1) : (t + 1);
            if (tn >= 0 && tn < SEQ && tid < 64)
                __builtin_prefetch(xg + (long)tn * G4H + tid * 32, 0, 1);
        }
        __syncthreads();

        // --- g[colbase..colbase+63] = h @ Whh^T  (WMMA, 4 indep chains) ---
        v8f a0, a1, a2, a3;
        #pragma unroll
        for (int r = 0; r < 8; ++r) { a0[r] = 0.f; a1[r] = 0.f; a2[r] = 0.f; a3[r] = 0.f; }

        #pragma unroll
        for (int k0 = 0; k0 < HID; k0 += 32) {
            v16bf a = load_a_frag(hb, k0, kg);             // LDS broadcast
            v16bf b0 = *(const v16bf*)(pb0 + k0);
            v16bf b1 = *(const v16bf*)(pb1 + k0);
            v16bf b2 = *(const v16bf*)(pb2 + k0);
            v16bf b3 = *(const v16bf*)(pb3 + k0);
            a0 = __builtin_amdgcn_wmma_f32_16x16x32_bf16(false, a, false, b0, (short)0, a0, false, false);
            a1 = __builtin_amdgcn_wmma_f32_16x16x32_bf16(false, a, false, b1, (short)0, a1, false, false);
            a2 = __builtin_amdgcn_wmma_f32_16x16x32_bf16(false, a, false, b2, (short)0, a2, false, false);
            a3 = __builtin_amdgcn_wmma_f32_16x16x32_bf16(false, a, false, b3, (short)0, a3, false, false);
        }

        // All 16 rows of D are identical (A rows were a broadcast of h):
        // lanes 0..15 element 0 carry out[col0 + lane]. Fold in xg[t].
        if (lane < 16) {
            const long xb = (long)t * G4H;
            g[colbase +  0 + lane] = a0[0] + xg[xb + colbase +  0 + lane];
            g[colbase + 16 + lane] = a1[0] + xg[xb + colbase + 16 + lane];
            g[colbase + 32 + lane] = a2[0] + xg[xb + colbase + 32 + lane];
            g[colbase + 48 + lane] = a3[0] + xg[xb + colbase + 48 + lane];
        }
        __syncthreads();

        // --- gates (PyTorch order i,f,g,o), state update ---
        if (tid < HID) {
            float ig = sigmoidf_fast(g[tid]);
            float fg = sigmoidf_fast(g[HID + tid]);
            float gg = tanhf(g[2 * HID + tid]);
            float og = sigmoidf_fast(g[3 * HID + tid]);
            c = fg * c + ig * gg;
            float h = og * tanhf(c);
            hf[tid] = h;
            hcat[(long)t * (2 * HID) + dir * HID + tid] = h;
        }
        __syncthreads();
    }
}

// ---------------------------------------------------------------------------
// 6) Output projection: feats[T,12] = hcat[T,1024] @ W_out^T + b_out
// ---------------------------------------------------------------------------
__global__ void k_proj(const float* __restrict__ hcat,
                       const float* __restrict__ Wout, const float* __restrict__ bout,
                       float* __restrict__ feats) {
    int i = blockIdx.x * blockDim.x + threadIdx.x;   // SEQ*TAGS
    if (i >= SEQ * TAGS) return;
    int t = i / TAGS, tag = i % TAGS;
    const float* h = hcat + (long)t * (2 * HID);
    const float* w = Wout + (long)tag * (2 * HID);
    float s = bout[tag];
    #pragma unroll 8
    for (int k = 0; k < 2 * HID; ++k) s += h[k] * w[k];
    feats[i] = s;
}

// ---------------------------------------------------------------------------
// 7) Viterbi: single wave. fv in LDS, backpointers to scratch, backtrack.
//    out[0] = score, out[1..SEQ] = path (as float, per output dtype).
// ---------------------------------------------------------------------------
__global__ __launch_bounds__(32)
void k_viterbi(const float* __restrict__ feats, const float* __restrict__ trans,
               int* __restrict__ bps, float* __restrict__ out) {
    __shared__ float fv[TAGS];
    const int tid = threadIdx.x;
    if (tid < TAGS) fv[tid] = (tid == START_TAG) ? 0.0f : NEGINF;
    __syncthreads();

    for (int t = 0; t < SEQ; ++t) {
        float best = NEGINF; int bp = 0;
        if (tid < TAGS) {
            best = fv[0] + trans[tid * TAGS + 0];
            for (int p = 1; p < TAGS; ++p) {
                float v = fv[p] + trans[tid * TAGS + p];
                if (v > best) { best = v; bp = p; }
            }
            best += feats[t * TAGS + tid];
            bps[t * TAGS + tid] = bp;
        }
        __syncthreads();
        if (tid < TAGS) fv[tid] = best;
        __syncthreads();
    }

    if (tid == 0) {
        float term[TAGS];
        for (int k = 0; k < TAGS; ++k) term[k] = fv[k] + trans[STOP_TAG * TAGS + k];
        term[STOP_TAG] = NEGINF;
        term[START_TAG] = NEGINF;
        int best = 0; float bs = term[0];
        for (int k = 1; k < TAGS; ++k) if (term[k] > bs) { bs = term[k]; best = k; }
        out[0] = bs;
        out[SEQ] = (float)best;                 // path[SEQ-1]
        int cur = best;
        for (int t = SEQ - 1; t >= 1; --t) {
            cur = bps[t * TAGS + cur];
            out[t] = (float)cur;                // path[t-1] at out[1+(t-1)]
        }
    }
}

// ---------------------------------------------------------------------------
// Host launcher
// ---------------------------------------------------------------------------
extern "C" void kernel_launch(void* const* d_in, const int* in_sizes, int n_in,
                              void* d_out, int out_size, void* d_ws, size_t ws_size,
                              hipStream_t stream) {
    (void)in_sizes; (void)n_in; (void)out_size; (void)ws_size;

    const int*   sentence = (const int*)  d_in[0];
    const float* table    = (const float*)d_in[1];
    const float* Wih_f    = (const float*)d_in[2];
    const float* Whh_f    = (const float*)d_in[3];
    const float* bih_f    = (const float*)d_in[4];
    const float* bhh_f    = (const float*)d_in[5];
    const float* Wih_b    = (const float*)d_in[6];
    const float* Whh_b    = (const float*)d_in[7];
    const float* bih_b    = (const float*)d_in[8];
    const float* bhh_b    = (const float*)d_in[9];
    const float* W_out    = (const float*)d_in[10];
    const float* b_out    = (const float*)d_in[11];
    const float* trans    = (const float*)d_in[12];
    float* out            = (float*)d_out;

    // scratch carving (256B aligned)
    char* ws = (char*)d_ws;
    size_t off = 0;
    auto take = [&](size_t bytes) -> void* {
        void* p = ws + off;
        off += (bytes + 255) & ~(size_t)255;
        return p;
    };
    bf16_t* emb    = (bf16_t*)take((size_t)SEQ * EPAD * 2);
    bf16_t* wihpf  = (bf16_t*)take((size_t)G4H * EPAD * 2);
    bf16_t* wihpb  = (bf16_t*)take((size_t)G4H * EPAD * 2);
    bf16_t* whhbf_f= (bf16_t*)take((size_t)G4H * HID * 2);
    bf16_t* whhbf_b= (bf16_t*)take((size_t)G4H * HID * 2);
    float*  biasf  = (float*) take((size_t)G4H * 4);
    float*  biasb  = (float*) take((size_t)G4H * 4);
    float*  xgf    = (float*) take((size_t)SEQ * G4H * 4);
    float*  xgb    = (float*) take((size_t)SEQ * G4H * 4);
    float*  hcat   = (float*) take((size_t)SEQ * 2 * HID * 4);
    float*  feats  = (float*) take((size_t)SEQ * TAGS * 4);
    int*    bps    = (int*)   take((size_t)SEQ * TAGS * 4);

    // 1) embedding gather -> bf16 padded
    k_embed<<<SEQ, EPAD, 0, stream>>>(sentence, table, emb);

    // 2) weight conversion (layout preserved: [N,K] == B-fragment order)
    {
        long tot = (long)G4H * EPAD;
        int blk = 256, grd = (int)((tot + blk - 1) / blk);
        k_cvt<<<grd, blk, 0, stream>>>(Wih_f, wihpf, G4H, EMBED, EPAD);
        k_cvt<<<grd, blk, 0, stream>>>(Wih_b, wihpb, G4H, EMBED, EPAD);
        long tot2 = (long)G4H * HID;
        int grd2 = (int)((tot2 + blk - 1) / blk);
        k_cvt<<<grd2, blk, 0, stream>>>(Whh_f, whhbf_f, G4H, HID, HID);
        k_cvt<<<grd2, blk, 0, stream>>>(Whh_b, whhbf_b, G4H, HID, HID);
    }

    // 3) fused biases
    k_bias<<<(G4H + 255) / 256, 256, 0, stream>>>(bih_f, bhh_f, biasf, G4H);
    k_bias<<<(G4H + 255) / 256, 256, 0, stream>>>(bih_b, bhh_b, biasb, G4H);

    // 4) input projection GEMMs (both directions), WMMA, 2x4 macro-tiles
    {
        dim3 grd(G4H / 256, SEQ / 32, 2);
        k_gemm_xg<<<grd, 128, 0, stream>>>(emb, wihpf, wihpb, biasf, biasb, xgf, xgb);
    }

    // 5) recurrent LSTM, forward + backward concurrently
    k_lstm<<<2, 1024, 0, stream>>>(xgf, xgb, whhbf_f, whhbf_b, hcat);

    // 6) projection to tag space
    k_proj<<<(SEQ * TAGS + 255) / 256, 256, 0, stream>>>(hcat, W_out, b_out, feats);

    // 7) Viterbi decode
    k_viterbi<<<1, 32, 0, stream>>>(feats, trans, bps, out);
}
